// HeteroGraphTransformer_19593640805065
// MI455X (gfx1250) — matrix operand order repaired and checked by reading, not compiled, not run on hardware
//
#include <hip/hip_runtime.h>
#include <hip/hip_bf16.h>
#include <math.h>

// ---- model constants (match reference) ----
#define DIN_   64
#define HID_   128
#define HEADS_ 8
#define DH_    128
#define INNER_ 1024
#define FF_    512

typedef __attribute__((ext_vector_type(16))) _Float16 v16h;
typedef __attribute__((ext_vector_type(8)))  _Float16 v8h;
typedef __attribute__((ext_vector_type(4)))  _Float16 h4;
typedef __attribute__((ext_vector_type(8)))  float    v8f;
typedef __attribute__((ext_vector_type(4)))  float    f4;

// Monotone float<->uint encoding so float max == unsigned max (for atomicMax).
__device__ __forceinline__ unsigned f32_ord(float f) {
  unsigned u = __float_as_uint(f);
  return (u & 0x80000000u) ? ~u : (u | 0x80000000u);
}
__device__ __forceinline__ float ord_f32(unsigned k) {
  unsigned u = (k & 0x80000000u) ? (k ^ 0x80000000u) : ~k;
  return __uint_as_float(u);
}

// ============================================================================
// One-shot conversion passes (hoisted out of the GEMM hot loop)
// ============================================================================

// fp32 -> f16, row-major, 4 elements/thread (b128 in, b64 out). n % 4 == 0.
__global__ void cvt_f16_kernel(const float* __restrict__ a, _Float16* __restrict__ o, int n4)
{
  const int i = blockIdx.x * blockDim.x + threadIdx.x;
  if (i >= n4) return;
  f4 v = ((const f4*)a)[i];
  h4 hv = { (_Float16)v[0], (_Float16)v[1], (_Float16)v[2], (_Float16)v[3] };
  ((h4*)o)[i] = hv;
}

// fp32 B[K,Nn] -> f16 fragment-order: Bs[(kt*ntiles + nt)*512 + lane*16 + j]
// so each wave-lane's 16-half B fragment is contiguous (two b128 loads).
__global__ void swizzleB_kernel(const float* __restrict__ B, _Float16* __restrict__ Bs,
                                int K, int Nn)
{
  const int i = blockIdx.x * blockDim.x + threadIdx.x;
  if (i >= K * Nn) return;
  const int j     = i & 15;
  const int lane  = (i >> 4) & 31;
  const int tile  = i >> 9;
  const int ntile = Nn >> 4;
  const int nt = tile % ntile, kt = tile / ntile;
  const int halfw = lane >> 4, l16 = lane & 15;
  const int kk = (kt << 5) + (halfw << 3) + (j & 7) + ((j >> 3) << 4);   // K index
  const int nn = (nt << 4) + l16;                // N index
  Bs[i] = (_Float16)B[(size_t)kk * Nn + nn];
}

// ============================================================================
// WMMA GEMM: C[M,Nn] = act(A16[M,K] @ B + bias) (+ addmat), f32 accumulate.
// One wave computes a 16x64 tile. Per k-step: A fragment (2xb128) + ALL four
// B fragments (8xb128, distinct registers) are loaded first, then 4 WMMAs
// issue back-to-back -> partial loadcnt waits, load latency hidden by WMMA.
// Optional f16 co-output C16 removes separate conversion passes.
// Requires: K%32==0, Nn%64==0 (true for all GEMMs here). M guarded per wave.
// ============================================================================
__global__ void wmma_gemm_f16_kernel(const _Float16* __restrict__ A16,
                                     const _Float16* __restrict__ Bs,
                                     const float* __restrict__ bias,
                                     const float* __restrict__ addmat,
                                     float* __restrict__ C, _Float16* __restrict__ C16,
                                     int M, int K, int Nn, int gelu)
{
  const int lane  = threadIdx.x & 31;
  const int wave  = threadIdx.x >> 5;
  const int halfw = lane >> 4;          // 0: lanes 0-15, 1: lanes 16-31
  const int l16   = lane & 15;
  const int m0 = (blockIdx.y * 4 + wave) * 16;
  const int n0 = blockIdx.x * 64;
  if (m0 >= M) return;                  // wave-uniform; EXEC all-ones for WMMA
  const int ntile = Nn >> 4;

  const v8f zero8 = {0.f, 0.f, 0.f, 0.f, 0.f, 0.f, 0.f, 0.f};
  v8f acc[4] = { zero8, zero8, zero8, zero8 };

  // A 16x32 fragment: lanes 0-15 row=l16 hold K 0..7 & 16..23; lanes 16-31 K 8..15 & 24..31
  const _Float16* Arow = A16 + (size_t)(m0 + l16) * K + halfw * 8;

  for (int k0 = 0; k0 < K; k0 += 32) {
    v8h alo = *(const v8h*)(Arow + k0);
    v8h ahi = *(const v8h*)(Arow + k0 + 16);

    const _Float16* bb = Bs + ((size_t)(k0 >> 5) * ntile + (n0 >> 4)) * 512 + lane * 16;
    v8h b0[4], b1[4];                   // distinct regs -> loads overlap WMMAs
    #pragma unroll
    for (int ss = 0; ss < 4; ++ss) {
      b0[ss] = *(const v8h*)(bb + ss * 512);
      b1[ss] = *(const v8h*)(bb + ss * 512 + 8);
    }

    v16h a;
    #pragma unroll
    for (int j = 0; j < 8; ++j) { a[j] = alo[j]; a[8 + j] = ahi[j]; }

    #pragma unroll
    for (int ss = 0; ss < 4; ++ss) {
      v16h b;
      #pragma unroll
      for (int j = 0; j < 8; ++j) { b[j] = b0[ss][j]; b[8 + j] = b1[ss][j]; }
      acc[ss] = __builtin_amdgcn_wmma_f32_16x16x32_f16(false, a, false, b, (short)0,
                                                       acc[ss], false, false);
    }
  }

  // C fragment: VGPR r -> row (halfw*8 + r), col = n0 + ss*16 + l16
  const int rbase = m0 + halfw * 8;
  #pragma unroll
  for (int ss = 0; ss < 4; ++ss) {
    const int   cg  = n0 + ss * 16 + l16;
    const float bvv = bias ? bias[cg] : 0.f;
    #pragma unroll
    for (int rr = 0; rr < 8; ++rr) {
      float val = acc[ss][rr] + bvv;
      if (gelu) val = 0.5f * val * (1.f + erff(val * 0.70710678118654752f)); // exact GELU
      size_t idx = (size_t)(rbase + rr) * Nn + cg;
      if (addmat) val += addmat[idx];
      C[idx] = val;
      if (C16) C16[idx] = (_Float16)val;  // fused f16 copy for downstream GEMMs
    }
  }
}

// ============================================================================
// Edge kernels: one wave per (edge, head), b128 loads, wave32 butterflies.
// ============================================================================
__global__ void edge_logits_kernel(const float* __restrict__ q, const float* __restrict__ k,
                                   const int* __restrict__ src, const int* __restrict__ dst,
                                   float* __restrict__ lg, unsigned* __restrict__ menc, int nwork)
{
  const int wid  = blockIdx.x * (blockDim.x >> 5) + (threadIdx.x >> 5);
  const int lane = threadIdx.x & 31;
  if (wid >= nwork) return;
  const int e = wid >> 3, h = wid & 7;
  const int s = src[e], d = dst[e];
  f4 qv = ((const f4*)(q + (size_t)d * INNER_ + h * DH_))[lane];
  f4 kv = ((const f4*)(k + (size_t)s * INNER_ + h * DH_))[lane];
  float sum = qv[0] * kv[0] + qv[1] * kv[1] + qv[2] * kv[2] + qv[3] * kv[3];
  #pragma unroll
  for (int off = 16; off > 0; off >>= 1) sum += __shfl_xor(sum, off, 32);
  sum *= 0.08838834764831845f; // 1/sqrt(128)
  if (lane == 0) {
    lg[wid] = sum;
    atomicMax(&menc[d * HEADS_ + h], f32_ord(sum)); // segment max
  }
}

__global__ void edge_expden_kernel(float* __restrict__ lg, const unsigned* __restrict__ menc,
                                   const int* __restrict__ dst, float* __restrict__ den, int nwork)
{
  const int i = blockIdx.x * blockDim.x + threadIdx.x;
  if (i >= nwork) return;
  const int e = i >> 3, h = i & 7;
  const int d = dst[e];
  const float m  = ord_f32(menc[d * HEADS_ + h]);
  const float ex = expf(lg[i] - m);
  lg[i] = ex; // in-place: logits buffer becomes exp buffer
  atomicAdd(&den[d * HEADS_ + h], ex);
}

__global__ void edge_agg_kernel(const float* __restrict__ lg, const float* __restrict__ den,
                                const float* __restrict__ v, const int* __restrict__ src,
                                const int* __restrict__ dst, float* __restrict__ agg, int nwork)
{
  const int wid  = blockIdx.x * (blockDim.x >> 5) + (threadIdx.x >> 5);
  const int lane = threadIdx.x & 31;
  if (wid >= nwork) return;
  const int e = wid >> 3, h = wid & 7;
  const int s = src[e], d = dst[e];
  const float alpha = lg[wid] / (den[d * HEADS_ + h] + 1e-16f);
  f4 vv = ((const f4*)(v + (size_t)s * INNER_ + h * DH_))[lane];
  vv *= alpha;
  float* ap = agg + (size_t)d * INNER_ + h * DH_ + lane * 4;
  #pragma unroll
  for (int c = 0; c < 4; ++c) atomicAdd(&ap[c], vv[c]);
}

// out = beta * x_r + (1-beta) * mean_heads(agg); beta = sigmoid([out,xr,out-xr] @ Wb)
__global__ void combine_kernel(const float* __restrict__ agg, const float* __restrict__ xr,
                               const float* __restrict__ Wb, float* __restrict__ outn, int N_)
{
  const int n    = blockIdx.x * (blockDim.x >> 5) + (threadIdx.x >> 5);
  const int lane = threadIdx.x & 31;
  if (n >= N_) return;
  f4 ov = {0.f, 0.f, 0.f, 0.f};
  #pragma unroll
  for (int h = 0; h < HEADS_; ++h)
    ov += ((const f4*)(agg + (size_t)n * INNER_ + h * DH_))[lane];
  ov *= (1.f / HEADS_);
  f4 xv = ((const f4*)(xr + (size_t)n * HID_))[lane];
  f4 w0 = ((const f4*)(Wb))[lane];
  f4 w1 = ((const f4*)(Wb + HID_))[lane];
  f4 w2 = ((const f4*)(Wb + 2 * HID_))[lane];
  f4 dv = ov * w0 + xv * w1 + (ov - xv) * w2;
  float dotp = dv[0] + dv[1] + dv[2] + dv[3];
  #pragma unroll
  for (int off = 16; off > 0; off >>= 1) dotp += __shfl_xor(dotp, off, 32);
  const float beta = 1.f / (1.f + expf(-dotp));
  f4 res = xv * beta + ov * (1.f - beta);
  ((f4*)(outn + (size_t)n * HID_))[lane] = res;
}

// r = LayerNorm(h + new) * g + b ; one wave per row of 128; co-writes f16 copy
__global__ void ln_add_kernel(const float* __restrict__ h, const float* __restrict__ nw,
                              const float* __restrict__ g, const float* __restrict__ b,
                              float* __restrict__ r, _Float16* __restrict__ r16, int N_)
{
  const int n    = blockIdx.x * (blockDim.x >> 5) + (threadIdx.x >> 5);
  const int lane = threadIdx.x & 31;
  if (n >= N_) return;
  f4 x = ((const f4*)(h + (size_t)n * HID_))[lane] + ((const f4*)(nw + (size_t)n * HID_))[lane];
  float s  = x[0] + x[1] + x[2] + x[3];
  float sq = x[0] * x[0] + x[1] * x[1] + x[2] * x[2] + x[3] * x[3];
  #pragma unroll
  for (int off = 16; off > 0; off >>= 1) {
    s  += __shfl_xor(s,  off, 32);
    sq += __shfl_xor(sq, off, 32);
  }
  const float mu  = s * (1.f / HID_);
  float var = sq * (1.f / HID_) - mu * mu;
  var = var < 0.f ? 0.f : var;
  const float rstd = rsqrtf(var + 1e-5f);
  f4 g4 = ((const f4*)g)[lane];
  f4 b4 = ((const f4*)b)[lane];
  f4 res = (x - mu) * rstd * g4 + b4;
  ((f4*)(r + (size_t)n * HID_))[lane] = res;
  h4 res16 = { (_Float16)res[0], (_Float16)res[1], (_Float16)res[2], (_Float16)res[3] };
  ((h4*)(r16 + (size_t)n * HID_))[lane] = res16;
}

__global__ void vadd_kernel(const float* __restrict__ a, const float* __restrict__ b,
                            float* __restrict__ c, int n)
{
  const int i = blockIdx.x * blockDim.x + threadIdx.x;
  if (i < n) c[i] = a[i] + b[i];
}

// ============================================================================
// Host-side orchestration helpers (all launches on `stream`; graph-capturable)
// ============================================================================
static void cvt16(const float* a, _Float16* o, size_t n, hipStream_t s)
{
  int n4 = (int)(n >> 2);
  cvt_f16_kernel<<<(n4 + 255) / 256, 256, 0, s>>>(a, o, n4);
}

static void gemm(const _Float16* A16, const float* B32, _Float16* Bs,
                 const float* bias, const float* addmat, float* C, _Float16* C16,
                 int M, int K, int Nn, int gelu, hipStream_t s)
{
  const int tot = K * Nn;
  swizzleB_kernel<<<(tot + 255) / 256, 256, 0, s>>>(B32, Bs, K, Nn);
  dim3 grid(Nn / 64, (M / 16 + 3) / 4);
  wmma_gemm_f16_kernel<<<grid, dim3(128), 0, s>>>(A16, Bs, bias, addmat, C, C16, M, K, Nn, gelu);
}

static void tconv(const _Float16* h16src, const _Float16* h16dst,
                  const int* src, const int* dst,
                  const float* Wq, const float* bq, const float* Wk, const float* bk,
                  const float* Wv, const float* bv, const float* Ws, const float* bs,
                  const float* Wb, float* out_new,
                  float* qb /*becomes agg*/, float* kb, float* vb, float* xr,
                  float* lg, float* den, unsigned* menc, _Float16* Bs,
                  int N_, int E_, hipStream_t s)
{
  gemm(h16dst, Wq, Bs, bq, nullptr, qb, nullptr, N_, HID_, INNER_, 0, s);
  gemm(h16src, Wk, Bs, bk, nullptr, kb, nullptr, N_, HID_, INNER_, 0, s);
  gemm(h16src, Wv, Bs, bv, nullptr, vb, nullptr, N_, HID_, INNER_, 0, s);
  gemm(h16dst, Ws, Bs, bs, nullptr, xr, nullptr, N_, HID_, HID_,  0, s);

  hipMemsetAsync(menc, 0, (size_t)N_ * HEADS_ * sizeof(unsigned), s);
  hipMemsetAsync(den,  0, (size_t)N_ * HEADS_ * sizeof(float),    s);

  const int nwork = E_ * HEADS_;                 // one wave per (edge, head)
  edge_logits_kernel<<<(nwork + 7) / 8, 256, 0, s>>>(qb, kb, src, dst, lg, menc, nwork);
  // q no longer needed -> reuse its 40MB buffer as the aggregation accumulator
  hipMemsetAsync(qb, 0, (size_t)N_ * INNER_ * sizeof(float), s);
  edge_expden_kernel<<<(nwork + 255) / 256, 256, 0, s>>>(lg, menc, dst, den, nwork);
  edge_agg_kernel<<<(nwork + 7) / 8, 256, 0, s>>>(lg, den, vb, src, dst, qb, nwork);
  combine_kernel<<<(N_ + 7) / 8, 256, 0, s>>>(qb, xr, Wb, out_new, N_);
}

static void ffn(float* h, _Float16* h16, const float* nw, const float* g, const float* b,
                const float* W1, const float* b1, const float* W2, const float* b2,
                float* r, float* t, _Float16* act16, _Float16* Bs, int N_, hipStream_t s)
{
  // act16 holds r16 then t16 (sequential uses on the same stream)
  ln_add_kernel<<<(N_ + 7) / 8, 256, 0, s>>>(h, nw, g, b, r, act16, N_);
  _Float16* t16 = act16;  // reuse: W1 GEMM reads r16 and writes t16 in the same launch, unordered across waves
  gemm(act16, W1, Bs, b1, nullptr, t, t16, N_, HID_, FF_, 1, s);  // GELU fused + f16 out
  gemm(t16,   W2, Bs, b2, r,       h, h16, N_, FF_, HID_, 0, s);  // h = r + gelu@W2 + b2
}

extern "C" void kernel_launch(void* const* d_in, const int* in_sizes, int n_in,
                              void* d_out, int out_size, void* d_ws, size_t ws_size,
                              hipStream_t stream)
{
  // ---- input pointers (setup_inputs dict order) ----
  const float* x_job    = (const float*)d_in[0];
  const float* Win_job  = (const float*)d_in[1];
  const float* bin_job  = (const float*)d_in[2];
  const float* pos_job  = (const float*)d_in[3];
  const float* te_job   = (const float*)d_in[4];
  const float* lng_job  = (const float*)d_in[5];
  const float* lnb_job  = (const float*)d_in[6];
  const float* W1_job   = (const float*)d_in[7];
  const float* b1_job   = (const float*)d_in[8];
  const float* W2_job   = (const float*)d_in[9];
  const float* b2_job   = (const float*)d_in[10];
  const float* Wout_job = (const float*)d_in[11];
  const float* bout_job = (const float*)d_in[12];
  const float* x_mach    = (const float*)d_in[13];
  const float* Win_mach  = (const float*)d_in[14];
  const float* bin_mach  = (const float*)d_in[15];
  const float* pos_mach  = (const float*)d_in[16];
  const float* te_mach   = (const float*)d_in[17];
  const float* lng_mach  = (const float*)d_in[18];
  const float* lnb_mach  = (const float*)d_in[19];
  const float* W1_mach   = (const float*)d_in[20];
  const float* b1_mach   = (const float*)d_in[21];
  const float* W2_mach   = (const float*)d_in[22];
  const float* b2_mach   = (const float*)d_in[23];
  const float* Wout_mach = (const float*)d_in[24];
  const float* bout_mach = (const float*)d_in[25];
  const float* Wq_jm = (const float*)d_in[26];
  const float* Wk_jm = (const float*)d_in[27];
  const float* Wv_jm = (const float*)d_in[28];
  const float* Ws_jm = (const float*)d_in[29];
  const float* Wb_jm = (const float*)d_in[30];
  const float* bq_jm = (const float*)d_in[31];
  const float* bk_jm = (const float*)d_in[32];
  const float* bv_jm = (const float*)d_in[33];
  const float* bs_jm = (const float*)d_in[34];
  const int*   ei_jm = (const int*)d_in[35];
  const float* Wq_mj = (const float*)d_in[36];
  const float* Wk_mj = (const float*)d_in[37];
  const float* Wv_mj = (const float*)d_in[38];
  const float* Ws_mj = (const float*)d_in[39];
  const float* Wb_mj = (const float*)d_in[40];
  const float* bq_mj = (const float*)d_in[41];
  const float* bk_mj = (const float*)d_in[42];
  const float* bv_mj = (const float*)d_in[43];
  const float* bs_mj = (const float*)d_in[44];
  const int*   ei_mj = (const int*)d_in[45];

  const int N_ = in_sizes[0] / DIN_;   // 10000 (divisible by 16)
  const int E_ = in_sizes[35] / 2;     // 60000

  // ---- workspace carve-out (256B-aligned slices) ----
  char* p = (char*)d_ws;
  auto takeB = [&](size_t bytes) {
    char* r = p;
    p += (bytes + 255) & ~(size_t)255;
    return r;
  };
  const size_t NH = (size_t)N_ * HID_;
  float* h_j   = (float*)takeB(NH * 4);
  float* h_m   = (float*)takeB(NH * 4);
  float* qb    = (float*)takeB((size_t)N_ * INNER_ * 4);   // q, later reused as agg
  float* kb    = (float*)takeB((size_t)N_ * INNER_ * 4);
  float* vb    = (float*)takeB((size_t)N_ * INNER_ * 4);
  float* xr    = (float*)takeB(NH * 4);
  float* new_j = (float*)takeB(NH * 4);
  float* new_m = (float*)takeB(NH * 4);
  float* rb    = (float*)takeB(NH * 4);
  float* tb    = (float*)takeB((size_t)N_ * FF_ * 4);
  float* lg    = (float*)takeB((size_t)E_ * HEADS_ * 4);
  float* den   = (float*)takeB((size_t)N_ * HEADS_ * 4);
  unsigned* menc = (unsigned*)takeB((size_t)N_ * HEADS_ * 4);
  float* biasc_j = (float*)takeB(HID_ * 4);
  float* biasc_m = (float*)takeB(HID_ * 4);
  _Float16* h16_j = (_Float16*)takeB(NH * 2);
  _Float16* h16_m = (_Float16*)takeB(NH * 2);
  _Float16* act16 = (_Float16*)takeB((size_t)N_ * FF_ * 2);     // x16 / r16 / t16 scratch
  _Float16* Bs    = (_Float16*)takeB((size_t)HID_ * INNER_ * 2); // largest weight, 256KB

  // ---- input projection: h = x @ Win + (bin + te) + pos (f16 copy fused) ----
  vadd_kernel<<<1, HID_, 0, stream>>>(bin_job,  te_job,  biasc_j, HID_);
  vadd_kernel<<<1, HID_, 0, stream>>>(bin_mach, te_mach, biasc_m, HID_);
  cvt16(x_job, act16, (size_t)N_ * DIN_, stream);
  gemm(act16, Win_job, Bs, biasc_j, pos_job, h_j, h16_j, N_, DIN_, HID_, 0, stream);
  cvt16(x_mach, act16, (size_t)N_ * DIN_, stream);
  gemm(act16, Win_mach, Bs, biasc_m, pos_mach, h_m, h16_m, N_, DIN_, HID_, 0, stream);

  // ---- layers ----
  for (int l = 0; l < 2; ++l) {
    const size_t oQ = (size_t)l * HID_ * INNER_, oS = (size_t)l * HID_ * HID_;
    const size_t oB = (size_t)l * 3 * HID_, oq = (size_t)l * INNER_, os = (size_t)l * HID_;
    // jm: job(src) -> mach(dst) : new_m   (both tconvs read pre-FFN h_j/h_m)
    tconv(h16_j, h16_m, ei_jm, ei_jm + E_,
          Wq_jm + oQ, bq_jm + oq, Wk_jm + oQ, bk_jm + oq, Wv_jm + oQ, bv_jm + oq,
          Ws_jm + oS, bs_jm + os, Wb_jm + oB, new_m,
          qb, kb, vb, xr, lg, den, menc, Bs, N_, E_, stream);
    // mj: mach(src) -> job(dst) : new_j
    tconv(h16_m, h16_j, ei_mj, ei_mj + E_,
          Wq_mj + oQ, bq_mj + oq, Wk_mj + oQ, bk_mj + oq, Wv_mj + oQ, bv_mj + oq,
          Ws_mj + oS, bs_mj + os, Wb_mj + oB, new_j,
          qb, kb, vb, xr, lg, den, menc, Bs, N_, E_, stream);

    ffn(h_j, h16_j, new_j, lng_job + os, lnb_job + os,
        W1_job + (size_t)l * HID_ * FF_, b1_job + (size_t)l * FF_,
        W2_job + (size_t)l * FF_ * HID_, b2_job + os, rb, tb, act16, Bs, N_, stream);
    ffn(h_m, h16_m, new_m, lng_mach + os, lnb_mach + os,
        W1_mach + (size_t)l * HID_ * FF_, b1_mach + (size_t)l * FF_,
        W2_mach + (size_t)l * FF_ * HID_, b2_mach + os, rb, tb, act16, Bs, N_, stream);
  }

  // ---- output heads: d_out = stack([h_j@Wout_job+b, h_m@Wout_mach+b]) ----
  float* out = (float*)d_out;
  gemm(h16_j, Wout_job,  Bs, bout_job,  nullptr, out,                     nullptr,
       N_, HID_, HID_, 0, stream);
  gemm(h16_m, Wout_mach, Bs, bout_mach, nullptr, out + (size_t)N_ * HID_, nullptr,
       N_, HID_, HID_, 0, stream);
}
